// GraphAttSurv_28286654611573
// MI455X (gfx1250) — compile-verified, matrix operand order checked by reading, not run on hardware
//
#include <hip/hip_runtime.h>
#include <hip/hip_bf16.h>

typedef __attribute__((ext_vector_type(16))) _Float16 v16h;
typedef __attribute__((ext_vector_type(8)))  _Float16 v8h;
typedef __attribute__((ext_vector_type(8)))  float    v8f;

#define N_NODES  50000
#define N_EDGES  800000
#define N_GRAPHS 8
#define WAVES    4

// ---------------------------------------------------------------------------
// float atomic max via int/uint trick (buffer must be initialized to -inf)
// ---------------------------------------------------------------------------
__device__ __forceinline__ void atomicMaxF(float* addr, float value) {
    if (value >= 0.0f)
        atomicMax((int*)addr, __float_as_int(value));
    else
        atomicMin((unsigned int*)addr, __float_as_uint(value));
}

// ---------------------------------------------------------------------------
// Pack fp32 weight [Kdim x Ncols] (row stride srcStride) into f16 WMMA
// B-fragments: tile = kt*NT + nt ; per tile 32 lanes x 16 halfs, lane holds
// column n = nt*16 + (lane&15); element i -> k = kt*32 + (i<8 ? hi*8+i
//                                                        : 16+hi*8+(i-8))
// ---------------------------------------------------------------------------
__global__ void frag_pack_kernel(const float* __restrict__ W,
                                 _Float16* __restrict__ dstf,
                                 int Kdim, int NT, int srcStride, int Ncols) {
    int tile = blockIdx.x;          // kt*NT + nt
    int kt = tile / NT, nt = tile % NT;
    int t = threadIdx.x;            // 0..511
    int lane = t >> 4, i = t & 15;
    int hi = lane >> 4, ln = lane & 15;
    int k = kt * 32 + (i < 8 ? hi * 8 + i : 16 + hi * 8 + (i - 8));
    int n = nt * 16 + ln;
    float v = (k < Kdim && n < Ncols) ? W[(size_t)k * srcStride + n] : 0.0f;
    dstf[((size_t)tile * 32 + lane) * 16 + i] = (_Float16)v;
}

// ---------------------------------------------------------------------------
// Per-layer node-state init
// ---------------------------------------------------------------------------
__global__ void node_init_kernel(float* __restrict__ nodemax,
                                 float* __restrict__ nodesum,
                                 float* __restrict__ hout, int n, int fout) {
    int i = blockIdx.x * blockDim.x + threadIdx.x;
    if (i < n) {
        nodemax[i] = -__builtin_inff();
        nodesum[i] = 0.0f;
        for (int c = 0; c < fout; ++c) hout[(size_t)i * fout + c] = 0.0f;
    }
}

// ---------------------------------------------------------------------------
// Edge MLP via WMMA. One wave = 16 edges, block = 4 waves = 64 edges.
//   feat (LDS, f16 [16 x KP]) -> GEMM1 -> relu -> h1 (LDS) -> GEMM2 -> relu
//   -> h2 (LDS, reuse feat) -> GEMM3 -> m, logits, atomicMax(nodemax[dst])
// ---------------------------------------------------------------------------
template <int FIN, int FOUT>
__global__ __launch_bounds__(WAVES * 32) void edge_mlp_kernel(
    const float* __restrict__ hin,          // [N, FIN]
    const int* __restrict__ src, const int* __restrict__ dst,
    const float* __restrict__ eattr,        // [E]
    const _Float16* __restrict__ w1f, const float* __restrict__ b1,
    const _Float16* __restrict__ w2f, const float* __restrict__ b2,
    const _Float16* __restrict__ w3f, const float* __restrict__ b3,
    const float* __restrict__ att,
    float* __restrict__ m_out,              // [E, FOUT]
    float* __restrict__ logit_out,          // [E]
    float* __restrict__ nodemax, int nEdges) {
    constexpr int K1  = 2 * FIN + 1;
    constexpr int K1T = (K1 + 31) / 32;
    constexpr int KP  = K1T * 32;
    constexpr int NT3 = (FOUT + 15) / 16;

    __shared__ __align__(16) _Float16 featLds[WAVES][16 * KP];
    __shared__ __align__(16) _Float16 h1Lds[WAVES][16 * 64];
    __shared__ float logitLds[WAVES][16];

    const int wave = threadIdx.x >> 5;
    const int lane = threadIdx.x & 31;
    const int hi   = lane >> 4;
    const int ln   = lane & 15;
    const int base = (blockIdx.x * WAVES + wave) * 16;

    _Float16* feat = featLds[wave];
    _Float16* h1   = h1Lds[wave];

    // ---- gather [x_src | x_dst | edge_attr | 0-pad] into LDS (f16) ----
    for (int idx = lane; idx < 16 * KP; idx += 32) {
        int row = idx / KP, col = idx - row * KP;
        int e = base + row; if (e >= nEdges) e = nEdges - 1;
        float v = 0.0f;
        if (col < FIN)            v = hin[(size_t)src[e] * FIN + col];
        else if (col < 2 * FIN)   v = hin[(size_t)dst[e] * FIN + (col - FIN)];
        else if (col == 2 * FIN)  v = eattr[e];
        feat[idx] = (_Float16)v;
    }
    if (lane < 16) logitLds[wave][lane] = 0.0f;
    __syncthreads();

    // ---- GEMM1: [16 x K1] x [K1 x 64] ----
    {
        v8f acc[4] = {};
#pragma unroll
        for (int kt = 0; kt < K1T; ++kt) {
            const _Float16* rp = feat + ln * KP + kt * 32;
            v8h a0 = *(const v8h*)(rp + hi * 8);
            v8h a1 = *(const v8h*)(rp + 16 + hi * 8);
            v16h a = __builtin_shufflevector(a0, a1, 0, 1, 2, 3, 4, 5, 6, 7,
                                             8, 9, 10, 11, 12, 13, 14, 15);
#pragma unroll
            for (int nt = 0; nt < 4; ++nt) {
                v16h b = *(const v16h*)(w1f + ((size_t)(kt * 4 + nt) * 32 + lane) * 16);
                acc[nt] = __builtin_amdgcn_wmma_f32_16x16x32_f16(
                    false, a, false, b, (short)0, acc[nt], false, false);
            }
        }
#pragma unroll
        for (int nt = 0; nt < 4; ++nt) {
            float bb = b1[nt * 16 + ln];
#pragma unroll
            for (int r = 0; r < 8; ++r) {
                float v = acc[nt][r] + bb;
                v = v > 0.0f ? v : 0.0f;                    // relu
                h1[(r + 8 * hi) * 64 + nt * 16 + ln] = (_Float16)v;
            }
        }
    }
    __syncthreads();

    // ---- GEMM2: [16 x 64] x [64 x 64] ----
    {
        v8f acc[4] = {};
#pragma unroll
        for (int kt = 0; kt < 2; ++kt) {
            const _Float16* rp = h1 + ln * 64 + kt * 32;
            v8h a0 = *(const v8h*)(rp + hi * 8);
            v8h a1 = *(const v8h*)(rp + 16 + hi * 8);
            v16h a = __builtin_shufflevector(a0, a1, 0, 1, 2, 3, 4, 5, 6, 7,
                                             8, 9, 10, 11, 12, 13, 14, 15);
#pragma unroll
            for (int nt = 0; nt < 4; ++nt) {
                v16h b = *(const v16h*)(w2f + ((size_t)(kt * 4 + nt) * 32 + lane) * 16);
                acc[nt] = __builtin_amdgcn_wmma_f32_16x16x32_f16(
                    false, a, false, b, (short)0, acc[nt], false, false);
            }
        }
#pragma unroll
        for (int nt = 0; nt < 4; ++nt) {
            float bb = b2[nt * 16 + ln];
#pragma unroll
            for (int r = 0; r < 8; ++r) {
                float v = acc[nt][r] + bb;
                v = v > 0.0f ? v : 0.0f;                    // relu
                feat[(r + 8 * hi) * 64 + nt * 16 + ln] = (_Float16)v;  // h2 reuses feat
            }
        }
    }
    __syncthreads();

    // ---- GEMM3: [16 x 64] x [64 x FOUT] + logits ----
    {
        v8f acc[NT3] = {};
#pragma unroll
        for (int kt = 0; kt < 2; ++kt) {
            const _Float16* rp = feat + ln * 64 + kt * 32;
            v8h a0 = *(const v8h*)(rp + hi * 8);
            v8h a1 = *(const v8h*)(rp + 16 + hi * 8);
            v16h a = __builtin_shufflevector(a0, a1, 0, 1, 2, 3, 4, 5, 6, 7,
                                             8, 9, 10, 11, 12, 13, 14, 15);
#pragma unroll
            for (int nt = 0; nt < NT3; ++nt) {
                v16h b = *(const v16h*)(w3f + ((size_t)(kt * NT3 + nt) * 32 + lane) * 16);
                acc[nt] = __builtin_amdgcn_wmma_f32_16x16x32_f16(
                    false, a, false, b, (short)0, acc[nt], false, false);
            }
        }
#pragma unroll
        for (int nt = 0; nt < NT3; ++nt) {
            int col = nt * 16 + ln;
            float bb = (col < FOUT) ? b3[col] : 0.0f;
            float av = (col < FOUT) ? att[col] : 0.0f;
#pragma unroll
            for (int r = 0; r < 8; ++r) {
                int M = r + 8 * hi;
                int e = base + M;
                float v = acc[nt][r] + bb;
                if (col < FOUT && e < nEdges)
                    m_out[(size_t)e * FOUT + col] = v;
                atomicAdd(&logitLds[wave][M], v * av);      // ds_add_f32
            }
        }
    }
    __syncthreads();

    if (lane < 16) {
        int e = base + lane;
        if (e < nEdges) {
            float lg = logitLds[wave][lane];
            lg = lg > 0.0f ? lg : 0.2f * lg;                // leaky_relu
            logit_out[e] = lg;
            atomicMaxF(&nodemax[dst[e]], lg);
        }
    }
}

// ---------------------------------------------------------------------------
// e = exp(logit - max[dst]);  denom[dst] += e
// ---------------------------------------------------------------------------
__global__ void edge_exp_kernel(const float* __restrict__ logit,
                                const int* __restrict__ dst,
                                const float* __restrict__ nodemax,
                                float* __restrict__ eexp,
                                float* __restrict__ nodesum, int nE) {
    int e = blockIdx.x * blockDim.x + threadIdx.x;
    if (e < nE) {
        int d = dst[e];
        float ex = __expf(logit[e] - nodemax[d]);
        eexp[e] = ex;
        atomicAdd(&nodesum[d], ex);
    }
}

// ---------------------------------------------------------------------------
// hout[dst, c] += (e / (denom[dst]+eps)) * m[e, c]   (thread per (edge,col))
// ---------------------------------------------------------------------------
__global__ void edge_agg_kernel(const float* __restrict__ m,
                                const float* __restrict__ eexp,
                                const float* __restrict__ nodesum,
                                const int* __restrict__ dst,
                                float* __restrict__ hout, int nE, int fout) {
    int idx = blockIdx.x * blockDim.x + threadIdx.x;   // < nE*fout (<2^31)
    if (idx < nE * fout) {
        int e = idx / fout;
        int c = idx - e * fout;
        int d = dst[e];
        float alpha = eexp[e] / (nodesum[d] + 1e-16f);
        atomicAdd(&hout[(size_t)d * fout + c], alpha * m[idx]);
    }
}

// ---------------------------------------------------------------------------
// Final pooling stages
// ---------------------------------------------------------------------------
__global__ void graph_init_kernel(float* gmax, float* gsum, float* Mpool) {
    int i = threadIdx.x;
    if (i < N_GRAPHS) { gmax[i] = -__builtin_inff(); gsum[i] = 0.0f; }
    if (i < N_GRAPHS * 8) Mpool[i] = 0.0f;
}

__global__ void node_att_kernel(const float* __restrict__ h,     // [N,8]
                                const float* __restrict__ Wa1,   // [8,128]
                                const float* __restrict__ ba1,   // [128]
                                const float* __restrict__ Wa2,   // [128]
                                const float* __restrict__ ba2,   // [1]
                                const int* __restrict__ batch,
                                float* __restrict__ a_buf, float* gmax, int n) {
    int i = blockIdx.x * blockDim.x + threadIdx.x;
    if (i < n) {
        float hv[8];
#pragma unroll
        for (int c = 0; c < 8; ++c) hv[c] = h[(size_t)i * 8 + c];
        float acc = ba2[0];
        for (int j = 0; j < 128; ++j) {
            float t = ba1[j];
#pragma unroll
            for (int c = 0; c < 8; ++c) t += hv[c] * Wa1[c * 128 + j];
            acc += tanhf(t) * Wa2[j];
        }
        a_buf[i] = acc;
        atomicMaxF(&gmax[batch[i]], acc);
    }
}

__global__ void node_exp_kernel(const float* __restrict__ a_buf,
                                const int* __restrict__ batch,
                                const float* __restrict__ gmax,
                                float* __restrict__ ea_buf, float* gsum, int n) {
    int i = blockIdx.x * blockDim.x + threadIdx.x;
    if (i < n) {
        int b = batch[i];
        float ea = __expf(a_buf[i] - gmax[b]);
        ea_buf[i] = ea;
        atomicAdd(&gsum[b], ea);
    }
}

__global__ void node_pool_kernel(const float* __restrict__ h,
                                 const float* __restrict__ ea_buf,
                                 const float* __restrict__ gsum,
                                 const int* __restrict__ batch,
                                 float* __restrict__ A_out,   // d_out + 8
                                 float* __restrict__ Mpool, int n) {
    int i = blockIdx.x * blockDim.x + threadIdx.x;
    if (i < n) {
        int b = batch[i];
        float A = ea_buf[i] / (gsum[b] + 1e-16f);
        A_out[i] = A;
#pragma unroll
        for (int c = 0; c < 8; ++c)
            atomicAdd(&Mpool[b * 8 + c], h[(size_t)i * 8 + c] * A);
    }
}

__global__ void final_kernel(const float* __restrict__ Mpool,
                             const float* __restrict__ Wc,
                             const float* __restrict__ bc,
                             float* __restrict__ Y_out) {
    int g = threadIdx.x;
    if (g < N_GRAPHS) {
        float y = bc[0];
#pragma unroll
        for (int c = 0; c < 8; ++c) y += Mpool[g * 8 + c] * Wc[c];
        Y_out[g] = 1.0f / (1.0f + __expf(-y));
    }
}

// ---------------------------------------------------------------------------
// Host launcher
// ---------------------------------------------------------------------------
extern "C" void kernel_launch(void* const* d_in, const int* in_sizes, int n_in,
                              void* d_out, int out_size, void* d_ws, size_t ws_size,
                              hipStream_t stream) {
    const int E = N_EDGES, N = N_NODES;
    const float* x      = (const float*)d_in[0];
    const int*   eidx   = (const int*)d_in[1];
    const float* eattr  = (const float*)d_in[2];
    const int*   batch  = (const int*)d_in[3];
    const int*   src    = eidx;
    const int*   dstp   = eidx + E;

    static const int FINs[4]  = {17, 64, 64, 64};
    static const int FOUTs[4] = {64, 64, 64, 8};

    // --- workspace carve (256B aligned bump allocator) ---
    char* p = (char*)d_ws;
    auto alloc = [&](size_t bytes) {
        void* r = (void*)p;
        p += (bytes + 255) & ~(size_t)255;
        return r;
    };
    float* hA      = (float*)alloc((size_t)N * 64 * 4);
    float* hB      = (float*)alloc((size_t)N * 64 * 4);
    float* m_buf   = (float*)alloc((size_t)E * 64 * 4);
    float* logit   = (float*)alloc((size_t)E * 4);
    float* eexp    = (float*)alloc((size_t)E * 4);
    float* nodemax = (float*)alloc((size_t)N * 4);
    float* nodesum = (float*)alloc((size_t)N * 4);
    float* a_buf   = (float*)alloc((size_t)N * 4);
    float* ea_buf  = (float*)alloc((size_t)N * 4);
    float* gmax    = (float*)alloc(N_GRAPHS * 4);
    float* gsum    = (float*)alloc(N_GRAPHS * 4);
    float* Mpool   = (float*)alloc(N_GRAPHS * 8 * 4);
    _Float16* w1f[4]; _Float16* w2f[4]; _Float16* w3f[4];
    for (int l = 0; l < 4; ++l) {
        int K1 = 2 * FINs[l] + 1, K1T = (K1 + 31) / 32;
        int NT3 = (FOUTs[l] + 15) / 16;
        w1f[l] = (_Float16*)alloc((size_t)K1T * 4 * 512 * 2);
        w2f[l] = (_Float16*)alloc((size_t)2 * 4 * 512 * 2);
        w3f[l] = (_Float16*)alloc((size_t)2 * NT3 * 512 * 2);
    }

    // --- pack weights into f16 WMMA fragments ---
    for (int l = 0; l < 4; ++l) {
        const float* W1 = (const float*)d_in[4 + 7 * l + 0];
        const float* W2 = (const float*)d_in[4 + 7 * l + 2];
        const float* W3 = (const float*)d_in[4 + 7 * l + 4];
        int K1 = 2 * FINs[l] + 1, K1T = (K1 + 31) / 32;
        int NT3 = (FOUTs[l] + 15) / 16;
        frag_pack_kernel<<<K1T * 4, 512, 0, stream>>>(W1, w1f[l], K1, 4, 64, 64);
        frag_pack_kernel<<<2 * 4, 512, 0, stream>>>(W2, w2f[l], 64, 4, 64, 64);
        frag_pack_kernel<<<2 * NT3, 512, 0, stream>>>(W3, w3f[l], 64, NT3, FOUTs[l], FOUTs[l]);
    }

    // --- 4 Edge-GAT layers ---
    const float* hin = x;
    float* hout = hA;
    const int edgeBlocks = E / (16 * WAVES);   // 12500 exact
    for (int l = 0; l < 4; ++l) {
        const float* b1  = (const float*)d_in[4 + 7 * l + 1];
        const float* b2  = (const float*)d_in[4 + 7 * l + 3];
        const float* b3  = (const float*)d_in[4 + 7 * l + 5];
        const float* att = (const float*)d_in[4 + 7 * l + 6];
        int fout = FOUTs[l];
        hout = (l & 1) ? hB : hA;

        node_init_kernel<<<(N + 255) / 256, 256, 0, stream>>>(nodemax, nodesum, hout, N, fout);

        if (l == 0)
            edge_mlp_kernel<17, 64><<<edgeBlocks, WAVES * 32, 0, stream>>>(
                hin, src, dstp, eattr, w1f[l], b1, w2f[l], b2, w3f[l], b3, att,
                m_buf, logit, nodemax, E);
        else if (l == 3)
            edge_mlp_kernel<64, 8><<<edgeBlocks, WAVES * 32, 0, stream>>>(
                hin, src, dstp, eattr, w1f[l], b1, w2f[l], b2, w3f[l], b3, att,
                m_buf, logit, nodemax, E);
        else
            edge_mlp_kernel<64, 64><<<edgeBlocks, WAVES * 32, 0, stream>>>(
                hin, src, dstp, eattr, w1f[l], b1, w2f[l], b2, w3f[l], b3, att,
                m_buf, logit, nodemax, E);

        edge_exp_kernel<<<(E + 255) / 256, 256, 0, stream>>>(logit, dstp, nodemax, eexp, nodesum, E);
        edge_agg_kernel<<<(E * fout + 255) / 256, 256, 0, stream>>>(m_buf, eexp, nodesum, dstp, hout, E, fout);
        hin = hout;
    }

    // --- attention pooling + survival head ---
    const float* Wa1 = (const float*)d_in[32];
    const float* ba1 = (const float*)d_in[33];
    const float* Wa2 = (const float*)d_in[34];
    const float* ba2 = (const float*)d_in[35];
    const float* Wc  = (const float*)d_in[36];
    const float* bc  = (const float*)d_in[37];
    float* out = (float*)d_out;

    graph_init_kernel<<<1, 128, 0, stream>>>(gmax, gsum, Mpool);
    node_att_kernel<<<(N + 127) / 128, 128, 0, stream>>>(hin, Wa1, ba1, Wa2, ba2, batch, a_buf, gmax, N);
    node_exp_kernel<<<(N + 255) / 256, 256, 0, stream>>>(a_buf, batch, gmax, ea_buf, gsum, N);
    node_pool_kernel<<<(N + 255) / 256, 256, 0, stream>>>(hin, ea_buf, gsum, batch, out + 8, Mpool, N);
    final_kernel<<<1, 32, 0, stream>>>(Mpool, Wc, bc, out);
}
